// DenseFeatureNumericEmbedding_38422777430071
// MI455X (gfx1250) — compile-verified, matrix operand order checked by reading, not run on hardware
//
#include <hip/hip_runtime.h>

typedef _Float16 v16h __attribute__((ext_vector_type(16)));
typedef float    v8f  __attribute__((ext_vector_type(8)));

#define BATCH 16384
#define NF 128
#define NH 128
#define NE 32
#define ROWS_PER_BLOCK 2048
#define WAVES 8
#define TILES (ROWS_PER_BLOCK / (WAVES * 16))      // 16 tiles of 16 rows per wave
#define BLOCKS_PER_F (BATCH / ROWS_PER_BLOCK)      // 8

__global__ __launch_bounds__(256)
void dfe_wmma_kernel(const float* __restrict__ x,
                     const float* __restrict__ w1,
                     const float* __restrict__ b1,
                     const float* __restrict__ w2,
                     const float* __restrict__ b2,
                     float* __restrict__ out)
{
    __shared__ float sW1[NH];
    __shared__ float sB1[NH];

    const int bx    = blockIdx.x;
    const int f     = bx / BLOCKS_PER_F;           // feature index
    const int chunk = bx % BLOCKS_PER_F;           // batch chunk
    const int tid   = threadIdx.x;
    const int wave  = tid >> 5;
    const int lane  = tid & 31;
    const int l15   = lane & 15;
    const int g     = lane >> 4;                   // lane half (A/B layout group)

    // Stage per-feature w1/b1 (f32) into LDS, block-cooperative.
    if (tid < NH)            sW1[tid]       = w1[f * NH + tid];
    else if (tid < 2 * NH)   sB1[tid - NH]  = b1[f * NH + (tid - NH)];
    __syncthreads();

    // ---------------------------------------------------------------
    // B fragments: w2[f] (H=128 x E=32) as f16, loaded once per wave
    // and reused across 16 batch tiles (w2[f] slice is L2-resident).
    // 16-bit B layout (K x N): lane holds column n = nt*16 + l15;
    // elements 0..7 -> K = ks*32 + g*8 + j ; 8..15 -> K = ks*32 + 16 + g*8 + (j-8)
    // ---------------------------------------------------------------
    v16h Bf[4][2];
    #pragma unroll
    for (int ks = 0; ks < 4; ++ks) {
        #pragma unroll
        for (int nt = 0; nt < 2; ++nt) {
            const int e = nt * 16 + l15;
            #pragma unroll
            for (int j = 0; j < 16; ++j) {
                const int k = ks * 32 + ((j < 8) ? (g * 8 + j) : (16 + g * 8 + (j - 8)));
                Bf[ks][nt][j] = (_Float16)w2[(f * NH + k) * NE + e];
            }
        }
    }

    const float b2v0 = b2[f * NE + l15];
    const float b2v1 = b2[f * NE + 16 + l15];

    const int chunk_base = chunk * ROWS_PER_BLOCK;

    for (int t = 0; t < TILES; ++t) {
        const int rowbase = chunk_base + (t * WAVES + wave) * 16;

        // x value for this lane's A-row (lanes L and L+16 share a row).
        const float xv = x[(size_t)(rowbase + l15) * NF + f];

        // Prefetch next tile's x column slice (gfx1250 global_prefetch).
        if (t + 1 < TILES) {
            const int nrb = chunk_base + ((t + 1) * WAVES + wave) * 16;
            __builtin_prefetch(&x[(size_t)(nrb + l15) * NF + f], 0, 0);
        }

        v8f acc0 = {};
        v8f acc1 = {};

        #pragma unroll
        for (int ks = 0; ks < 4; ++ks) {
            // Build A fragment in registers in the native 16-bit A layout:
            // h computed in f32 (exact fma), rounded once to f16 (2^-12 rel).
            v16h A;
            #pragma unroll
            for (int j = 0; j < 16; ++j) {
                const int h = ks * 32 + ((j < 8) ? (g * 8 + j) : (16 + g * 8 + (j - 8)));
                A[j] = (_Float16)fmaxf(fmaf(xv, sW1[h], sB1[h]), 0.0f);
            }
            acc0 = __builtin_amdgcn_wmma_f32_16x16x32_f16(false, A, false, Bf[ks][0], (short)0, acc0, false, false);
            acc1 = __builtin_amdgcn_wmma_f32_16x16x32_f16(false, A, false, Bf[ks][1], (short)0, acc1, false, false);
        }

        // C/D layout: VGPR i -> (m = i + 8*g, n = l15). Add b2 in f32.
        // Output is a 256MB pure stream (> 192MB L2): store non-temporal so
        // L2 stays reserved for the reused x / w2 operands.
        #pragma unroll
        for (int i = 0; i < 8; ++i) {
            const int m = i + g * 8;
            float* orow = out + (size_t)(rowbase + m) * (NF * NE) + f * NE;
            __builtin_nontemporal_store(acc0[i] + b2v0, &orow[l15]);
            __builtin_nontemporal_store(acc1[i] + b2v1, &orow[16 + l15]);
        }
    }
}

extern "C" void kernel_launch(void* const* d_in, const int* in_sizes, int n_in,
                              void* d_out, int out_size, void* d_ws, size_t ws_size,
                              hipStream_t stream) {
    const float* x  = (const float*)d_in[0];
    const float* w1 = (const float*)d_in[1];
    const float* b1 = (const float*)d_in[2];
    const float* w2 = (const float*)d_in[3];
    const float* b2 = (const float*)d_in[4];
    float* out = (float*)d_out;

    dim3 grid(NF * BLOCKS_PER_F);   // 1024 blocks, one feature per 8 blocks
    dim3 block(256);                // 8 waves (wave32)
    hipLaunchKernelGGL(dfe_wmma_kernel, grid, block, 0, stream,
                       x, w1, b1, w2, b2, out);
}
